// UsadModel_52020643889322
// MI455X (gfx1250) — matrix operand order repaired and checked by reading, not compiled, not run on hardware
//
#include <hip/hip_runtime.h>
#include <hip/hip_bf16.h>
#include <math.h>

typedef __bf16 bf16;
typedef __attribute__((ext_vector_type(16))) __bf16 v16bf;
typedef __attribute__((ext_vector_type(8)))  __bf16 v8bf;
typedef __attribute__((ext_vector_type(8)))  float  v8f;

namespace {
constexpr int H_  = 100;   // hidden
constexpr int V_  = 64;    // var length
constexpr int B_  = 512;   // batch
constexpr int S_  = 256;   // sequence
constexpr int HP_ = 128;   // padded hidden (8 x 16)
}

// ---------------------------------------------------------------------------
// Weight pre-pack kernels: convert fp32 PyTorch-layout weights into bf16
// WMMA B-fragment order. Fragment layout used here, contiguous k (16-bit B, 32x16):
//   lane l: column n = l%16 ; element e: k = (l/16)*16 + e   (within 32-K tile)
// Pack index: (((gate*8 + utile)*KK + kk)*32 + lane)*16 + e
// ---------------------------------------------------------------------------
__global__ void pack_w_kernel(const float* __restrict__ W, bf16* __restrict__ out,
                              int in_dim, int KK) {
  int id = blockIdx.x * blockDim.x + threadIdx.x;
  int total = 4 * 8 * KK * 512;
  if (id >= total) return;
  int e  = id & 15;
  int l  = (id >> 4) & 31;
  int kk = (id >> 9) % KK;
  int t  = (id / (512 * KK)) & 7;
  int j  = id / (512 * KK * 8);
  int u  = t * 16 + (l & 15);
  int k  = kk * 32 + (l >> 4) * 16 + e;
  float v = 0.0f;
  if (u < H_ && k < in_dim) v = W[(j * H_ + u) * in_dim + k];
  out[id] = (bf16)v;
}

__global__ void pack_fc_kernel(const float* __restrict__ W, bf16* __restrict__ out) {
  int id = blockIdx.x * blockDim.x + threadIdx.x;
  if (id >= 4 * 4 * 512) return;
  int e  = id & 15;
  int l  = (id >> 4) & 31;
  int kk = (id >> 9) & 3;
  int nt = id >> 11;
  int n  = nt * 16 + (l & 15);          // n < 64 always
  int k  = kk * 32 + (l >> 4) * 16 + e; // K padded 100 -> 128
  float v = (k < H_) ? W[n * H_ + k] : 0.0f;
  out[id] = (bf16)v;
}

__global__ void pack_bias_kernel(const float* __restrict__ bih,
                                 const float* __restrict__ bhh,
                                 float* __restrict__ out) {
  int id = blockIdx.x * blockDim.x + threadIdx.x;
  if (id >= 512) return;
  int j = id >> 7, u = id & 127;
  out[id] = (u < H_) ? (bih[j * H_ + u] + bhh[j * H_ + u]) : 0.0f;
}

// ---------------------------------------------------------------------------
// Device helpers
// ---------------------------------------------------------------------------
__device__ __forceinline__ v16bf load_a_frag(const bf16* buf, int stride, int kk, int lane) {
  // 16-bit A-matrix 16x32: lane m=l%16, hi=l/16; e -> k=kk*32+hi*16+e
  // (matches packed B order). Two contiguous 16B chunks.
  int m = lane & 15, hi = lane >> 4;
  const bf16* p = buf + m * stride + kk * 32 + hi * 16;
  v8bf c0 = *(const v8bf*)p;
  v8bf c1 = *(const v8bf*)(p + 8);
  v16bf a;
#pragma unroll
  for (int i = 0; i < 8; ++i) { a[i] = c0[i]; a[i + 8] = c1[i]; }
  return a;
}

__device__ __forceinline__ v16bf load_b_frag(const bf16* pack, int frag, int lane) {
  return *(const v16bf*)(pack + (size_t)frag * 512 + lane * 16);
}

__device__ __forceinline__ float sigmoidf_(float x) {
  return 1.0f / (1.0f + __expf(-x));
}

#define WMMA_BF16(A, Bf, C)                                                  \
  __builtin_amdgcn_wmma_f32_16x16x32_bf16(false, (A), false, (Bf), (short)0, \
                                          (C), false, false)

struct LayerP { const bf16* wih; const bf16* whh; const float* bias; };

struct UsadParams {
  const float* batch;
  LayerP enc0, enc1, d10, d11, d20, d21;
  const bf16* fc1; const float* fcb1;
  const bf16* fc2; const float* fcb2;
  float* errbuf;   // [32 blocks][32 lanes]
};

// One LSTM cell for the whole block. Waves 0..6 each own u-tile t=wave
// (u = t*16 + lane%16, valid u < 100; padded region stays exactly zero).
// Wave 7 just participates in the barriers.
template <int KKIN>
__device__ __forceinline__ void lstm_cell_block(const bf16* inbuf, int instride,
                                                const LayerP& P,
                                                bf16* hslot, float* cslot) {
  int tid = threadIdx.x;
  int wv = tid >> 5, lane = tid & 31;
  bool act = (wv < 7);
  float hn[8], cn[8];
  if (act) {
    int t = wv;
    v8f ai = {}, af = {}, ag = {}, ao = {};
#pragma unroll
    for (int kk = 0; kk < KKIN; ++kk) {           // x @ Wih^T
      v16bf a = load_a_frag(inbuf, instride, kk, lane);
      ai = WMMA_BF16(a, load_b_frag(P.wih, (0 * 8 + t) * KKIN + kk, lane), ai);
      af = WMMA_BF16(a, load_b_frag(P.wih, (1 * 8 + t) * KKIN + kk, lane), af);
      ag = WMMA_BF16(a, load_b_frag(P.wih, (2 * 8 + t) * KKIN + kk, lane), ag);
      ao = WMMA_BF16(a, load_b_frag(P.wih, (3 * 8 + t) * KKIN + kk, lane), ao);
    }
#pragma unroll
    for (int kk = 0; kk < 4; ++kk) {              // h @ Whh^T  (K padded 128)
      v16bf a = load_a_frag(hslot, HP_, kk, lane);
      ai = WMMA_BF16(a, load_b_frag(P.whh, (0 * 8 + t) * 4 + kk, lane), ai);
      af = WMMA_BF16(a, load_b_frag(P.whh, (1 * 8 + t) * 4 + kk, lane), af);
      ag = WMMA_BF16(a, load_b_frag(P.whh, (2 * 8 + t) * 4 + kk, lane), ag);
      ao = WMMA_BF16(a, load_b_frag(P.whh, (3 * 8 + t) * 4 + kk, lane), ao);
    }
    int u  = t * 16 + (lane & 15);
    int hi = lane >> 4;
    float bI = P.bias[u],        bF = P.bias[128 + u];
    float bG = P.bias[256 + u],  bO = P.bias[384 + u];
#pragma unroll
    for (int r = 0; r < 8; ++r) {                 // D tile: m = r + 8*hi, n = u
      int m = r + 8 * hi;
      float c  = cslot[m * HP_ + u];
      float gi = sigmoidf_(ai[r] + bI);
      float gf = sigmoidf_(af[r] + bF);
      float gg = tanhf(ag[r] + bG);
      float go = sigmoidf_(ao[r] + bO);
      float cnew = gf * c + gi * gg;
      cn[r] = cnew;
      hn[r] = go * tanhf(cnew);
    }
  }
  __syncthreads();   // all waves finished reading hslot
  if (act) {
    int t = wv;
    int u  = t * 16 + (lane & 15);
    int hi = lane >> 4;
#pragma unroll
    for (int r = 0; r < 8; ++r) {
      int m = r + 8 * hi;
      hslot[m * HP_ + u] = (bf16)hn[r];
      cslot[m * HP_ + u] = cn[r];
    }
  }
  __syncthreads();   // h/c updates visible to everyone
}

// FC projection (wave 7 only): pred = h1 @ fcW^T + b  -> x buffer (bf16),
// optionally accumulates squared error vs. the time-reversed batch slice.
__device__ __forceinline__ float fc_block(const bf16* hslot, const bf16* fcP,
                                          const float* fcb, bf16* xout,
                                          const float* cmp) {
  int tid = threadIdx.x;
  int wv = tid >> 5, lane = tid & 31;
  float errsum = 0.0f;
  if (wv == 7) {
    int hi = lane >> 4;
    v16bf afr[4];
#pragma unroll
    for (int kk = 0; kk < 4; ++kk) afr[kk] = load_a_frag(hslot, HP_, kk, lane);
#pragma unroll
    for (int nt = 0; nt < 4; ++nt) {
      v8f acc = {};
#pragma unroll
      for (int kk = 0; kk < 4; ++kk)
        acc = WMMA_BF16(afr[kk], load_b_frag(fcP, nt * 4 + kk, lane), acc);
      int n = nt * 16 + (lane & 15);
      float bb = fcb[n];
#pragma unroll
      for (int r = 0; r < 8; ++r) {
        int m = r + 8 * hi;
        float p = acc[r] + bb;
        xout[m * V_ + n] = (bf16)p;
        if (cmp) {
          float d = p - cmp[(size_t)m * (S_ * V_) + n];
          errsum += d * d;
        }
      }
    }
  }
  __syncthreads();
  return errsum;
}

// ---------------------------------------------------------------------------
// Main persistent kernel: one block = 16 batch rows, 8 waves, runs the whole
// encoder + decoder recurrence with h/c state resident in LDS.
// ---------------------------------------------------------------------------
__global__ void __launch_bounds__(256) usad_main_kernel(UsadParams p) {
  __shared__ alignas(32) bf16  h0[16 * HP_];
  __shared__ alignas(32) bf16  h1[16 * HP_];
  __shared__ alignas(32) float c0[16 * HP_];
  __shared__ alignas(32) float c1[16 * HP_];
  __shared__ alignas(32) bf16  xb[16 * V_];

  int tid = threadIdx.x;
  int rowBase = blockIdx.x * 16;

  for (int i = tid; i < 16 * HP_; i += 256) {
    h0[i] = (bf16)0.0f; h1[i] = (bf16)0.0f;
    c0[i] = 0.0f;       c1[i] = 0.0f;
  }
  for (int i = tid; i < 16 * V_; i += 256) xb[i] = (bf16)0.0f;
  __syncthreads();

  // ---- Encoder: 2-layer LSTM over the sequence ----
#pragma unroll 1
  for (int s = 0; s < S_; ++s) {
    for (int i = tid; i < 16 * V_; i += 256) {
      int m = i >> 6, n = i & 63;
      xb[i] = (bf16)p.batch[(size_t)(rowBase + m) * (S_ * V_) + s * V_ + n];
    }
    __syncthreads();
    lstm_cell_block<2>(xb, V_,  p.enc0, h0, c0);
    lstm_cell_block<4>(h0, HP_, p.enc1, h1, c1);
  }

  // ---- Decoder: x0 = 0, state carried from encoder ----
  for (int i = tid; i < 16 * V_; i += 256) xb[i] = (bf16)0.0f;
  __syncthreads();

  float err = 0.0f;
#pragma unroll 1
  for (int s = 0; s < S_; ++s) {
    lstm_cell_block<2>(xb, V_,  p.d10, h0, c0);
    lstm_cell_block<4>(h0, HP_, p.d11, h1, c1);
    fc_block(h1, p.fc1, p.fcb1, xb, nullptr);                 // x1
    lstm_cell_block<2>(xb, V_,  p.d20, h0, c0);
    lstm_cell_block<4>(h0, HP_, p.d21, h1, c1);
    const float* cmp = p.batch + (size_t)rowBase * (S_ * V_)
                               + (size_t)(S_ - 1 - s) * V_;   // time-reversed
    err += fc_block(h1, p.fc2, p.fcb2, xb, cmp);              // x2 + loss
  }

  if ((tid >> 5) == 7)
    p.errbuf[blockIdx.x * 32 + (tid & 31)] = err;   // deterministic partials
}

__global__ void finalize_kernel(const float* __restrict__ errbuf,
                                const int* __restrict__ n,
                                float* __restrict__ out) {
  if (threadIdx.x == 0 && blockIdx.x == 0) {
    float s = 0.0f;
    for (int i = 0; i < 32 * 32; ++i) s += errbuf[i];
    float m = s * (1.0f / (float)(B_ * S_ * V_));
    float inv_n = 1.0f / (float)n[0];
    out[0] = inv_n * m + (1.0f - inv_n) * m;
    out[1] = inv_n * m - (1.0f - inv_n) * m;
  }
}

// ---------------------------------------------------------------------------
// Host launcher
// ---------------------------------------------------------------------------
extern "C" void kernel_launch(void* const* d_in, const int* in_sizes, int n_in,
                              void* d_out, int out_size, void* d_ws, size_t ws_size,
                              hipStream_t stream) {
  (void)in_sizes; (void)n_in; (void)out_size; (void)ws_size;
  const float* batch = (const float*)d_in[0];
  const int*   n_ptr = (const int*)d_in[29];

  char* ws = (char*)d_ws;
  size_t off = 0;
  auto alloc = [&](size_t bytes) -> char* {
    char* ptr = ws + off;
    off += (bytes + 255) & ~(size_t)255;
    return ptr;
  };

  float* errbuf = (float*)alloc(32 * 32 * sizeof(float));

  struct LIdx { int wih, whh, bih, bhh, in; };
  const LIdx L[6] = {
    {1, 2, 3, 4, 64}, {5, 6, 7, 8, 100},          // encoder l0, l1
    {9, 10, 11, 12, 64}, {13, 14, 15, 16, 100},   // dec1 l0, l1
    {19, 20, 21, 22, 64}, {23, 24, 25, 26, 100},  // dec2 l0, l1
  };
  bf16* wihP[6]; bf16* whhP[6]; float* biasP[6];
  for (int i = 0; i < 6; ++i) {
    int KK = (L[i].in == 64) ? 2 : 4;
    size_t wihElems = (size_t)4 * 8 * KK * 512;
    size_t whhElems = (size_t)4 * 8 * 4 * 512;
    wihP[i]  = (bf16*)alloc(wihElems * sizeof(bf16));
    whhP[i]  = (bf16*)alloc(whhElems * sizeof(bf16));
    biasP[i] = (float*)alloc(512 * sizeof(float));
    pack_w_kernel<<<(int)((wihElems + 255) / 256), 256, 0, stream>>>(
        (const float*)d_in[L[i].wih], wihP[i], L[i].in, KK);
    pack_w_kernel<<<(int)((whhElems + 255) / 256), 256, 0, stream>>>(
        (const float*)d_in[L[i].whh], whhP[i], 100, 4);
    pack_bias_kernel<<<2, 256, 0, stream>>>(
        (const float*)d_in[L[i].bih], (const float*)d_in[L[i].bhh], biasP[i]);
  }
  bf16* fc1P = (bf16*)alloc((size_t)4 * 4 * 512 * sizeof(bf16));
  bf16* fc2P = (bf16*)alloc((size_t)4 * 4 * 512 * sizeof(bf16));
  pack_fc_kernel<<<(4 * 4 * 512) / 256, 256, 0, stream>>>((const float*)d_in[17], fc1P);
  pack_fc_kernel<<<(4 * 4 * 512) / 256, 256, 0, stream>>>((const float*)d_in[27], fc2P);

  UsadParams P;
  P.batch = batch;
  P.enc0 = {wihP[0], whhP[0], biasP[0]};
  P.enc1 = {wihP[1], whhP[1], biasP[1]};
  P.d10  = {wihP[2], whhP[2], biasP[2]};
  P.d11  = {wihP[3], whhP[3], biasP[3]};
  P.d20  = {wihP[4], whhP[4], biasP[4]};
  P.d21  = {wihP[5], whhP[5], biasP[5]};
  P.fc1 = fc1P; P.fcb1 = (const float*)d_in[18];
  P.fc2 = fc2P; P.fcb2 = (const float*)d_in[28];
  P.errbuf = errbuf;

  usad_main_kernel<<<32, 256, 0, stream>>>(P);
  finalize_kernel<<<1, 1, 0, stream>>>(errbuf, n_ptr, (float*)d_out);
}
